// TraceUnitFactorizedState_14963666059483
// MI455X (gfx1250) — matrix operand order repaired and checked by reading, not compiled, not run on hardware
//
#include <hip/hip_runtime.h>
#include <stdint.h>

// ---------------------------------------------------------------------------
// TraceUnitFactorizedState — MI455X (gfx1250) implementation.
// bf16 WMMA (v_wmma_f32_16x16x32_bf16) for all GEMMs, f32 accumulate.
// Weight matrices are pre-transposed AND k-pair-interleaved in global memory,
// so B-tile staging is a contiguous 16B-aligned copy -> CDNA5 async
// global->LDS instructions (ASYNCcnt), double-buffered in the GRU scan.
// Every WMMA fragment load is an aligned ds b32 (pairs merged to 2addr).
// ---------------------------------------------------------------------------

typedef __bf16 bf16;
typedef __attribute__((ext_vector_type(2)))  __bf16   bf16x2;
typedef __attribute__((ext_vector_type(16))) __bf16   bf16x16;
typedef __attribute__((ext_vector_type(8)))  float    f32x8;
typedef __attribute__((ext_vector_type(8)))  uint32_t u32x8;

#define WMMA_BF16(acc, a, b)                                                   \
  acc = __builtin_amdgcn_wmma_f32_16x16x32_bf16(false, a, false, b, (short)0,  \
                                                acc, false, false)

// Async global->LDS (gfx1250). Guarded: fall back to uint4 copies if the
// toolchain lacks the builtin. Builtin takes (v4i AS1* src, v4i AS3* dst,
// imm offset, imm cpol) per the clang diagnostic.
typedef int v4i_ __attribute__((vector_size(16)));
typedef __attribute__((address_space(1))) v4i_* as1_v4i;
typedef __attribute__((address_space(3))) v4i_* as3_v4i;

#if __has_builtin(__builtin_amdgcn_global_load_async_to_lds_b128)
  #define HAVE_ASYNC_LDS 1
  #if __has_builtin(__builtin_amdgcn_s_wait_asynccnt)
    #define WAIT_ASYNC(n) __builtin_amdgcn_s_wait_asynccnt(n)
  #else
    #define WAIT_ASYNC(n) asm volatile("s_wait_asynccnt %0" ::"i"(n) : "memory")
  #endif
#else
  #define HAVE_ASYNC_LDS 0
  #define WAIT_ASYNC(n) do {} while (0)
#endif

__device__ inline void async_copy16(bf16* lds_dst, const bf16* gsrc) {
#if HAVE_ASYNC_LDS
  __builtin_amdgcn_global_load_async_to_lds_b128(
      (as1_v4i)(gsrc), (as3_v4i)(lds_dst), 0, 0);
#else
  *(uint4*)lds_dst = *(const uint4*)gsrc;
#endif
}

// Problem dims
#define BB   8
#define TT   256
#define KK   64
#define MM   32
#define DD   384
#define D3   1152
#define BT   2048    // B*T
#define ROWS 65536   // B*T*M

// Output offsets (floats), reference tuple order
#define OFF_UNIT  0
#define OFF_PRES  25165824
#define OFF_ZDYN  25231360
#define OFF_ZSEM  50397184
#define OFF_ALPHA 75563008

// ---------------------------------------------------------------------------
// Fragment loaders — gfx1250 WMMA VGPR layouts (ISA 7.12.2).
// A (16x32 MxK, row-major LDS): lanes 0-15: M=lane, K {0..7,16..23};
//   lanes 16-31: M=lane-16, K {8..15,24..31}; K pairs contiguous -> b32 loads.
// B (32x16 KxN): lane group g: VGPR v holds K=8g+2v,+1 (v<4), K=8g+2v+8,+1 (v>=4),
//   N=lane&15. LDS/global hold B k-pair interleaved:
//   (k,n) -> ((k>>1)*S + n)*2 + (k&1), so each VGPR is one aligned b32 load.
// C/D: f32x8, VGPR q: M = q + 8*(lane>>4), N = lane&15.
// ---------------------------------------------------------------------------
__device__ inline bf16x16 load_a_frag(const bf16* sm, int stride, int lane) {
  int r = lane & 15, hi = lane >> 4;
  const bf16* row = sm + r * stride + hi * 8;
  u32x8 a;
#pragma unroll
  for (int v = 0; v < 8; ++v) {
    int k = (v < 4) ? 2 * v : 2 * v + 8;                  // even
    a[v] = *(const uint32_t*)(row + k);
  }
  return __builtin_bit_cast(bf16x16, a);
}

__device__ inline bf16x16 load_b_frag_i(const bf16* base, int stride, int lane) {
  int n = lane & 15, g = lane >> 4;
  u32x8 b;
#pragma unroll
  for (int v = 0; v < 8; ++v)
    b[v] = *(const uint32_t*)(base + ((4 * g + v + (v < 4 ? 0 : 4)) * stride + n) * 2);
  return __builtin_bit_cast(bf16x16, b);
}

__device__ inline float sigmoidf_(float x) { return 1.f / (1.f + __expf(-x)); }

// ---------------------------------------------------------------------------
// Weight transpose + convert + k-pair interleave:
//   src (J x 384) f32, element src[j][d];  d is the GEMM K dimension.
//   dst element (k=d, n=j) at ((d>>1)*J + j)*2 + (d&1), bf16.
// ---------------------------------------------------------------------------
__global__ void k_wtranspose_i(const float* __restrict__ src, bf16* __restrict__ dst, int J) {
  int i = blockIdx.x * 256 + threadIdx.x;
  if (i >= J * DD) return;
  int j = i / DD, d = i - j * DD;
  dst[((d >> 1) * J + j) * 2 + (d & 1)] = (bf16)src[i];
}

// ---------------------------------------------------------------------------
// Kernel 1: per (b,t) block.
//   U0[m,d] = (sum_k a[k,m] * x[k,d]) * rcp(max(sum_k a[k,m], 1e-6))
// 2 m-tiles x 24 n-tiles, K=64 (2 WMMA k-chunks). Writes U0 into the
// unit_inputs output region (projected in place later) + presence.
// ---------------------------------------------------------------------------
__global__ __launch_bounds__(256) void k_unit_pool(const float* __restrict__ tf,
                                                   const float* __restrict__ assign,
                                                   float* __restrict__ outU,
                                                   float* __restrict__ outPres) {
  __shared__ alignas(16) bf16 At[MM * KK];   // assignment^T : [m][k] row-major
  __shared__ alignas(16) bf16 Xs[KK * DD];   // token features, k-pair interleaved
  __shared__ float rden[MM];                 // 1 / denom
  int bt = blockIdx.x, tid = threadIdx.x;
  const float* ab = assign + (size_t)bt * KK * MM;
  const float* xb = tf + (size_t)bt * KK * DD;

  for (int e = tid; e < KK * MM; e += 256) {
    int k = e >> 5, m = e & 31;
    At[m * KK + k] = (bf16)ab[e];
  }
  for (int e = tid; e < KK * DD; e += 256) {
    int k = e / DD, d = e - k * DD;
    Xs[((k >> 1) * DD + d) * 2 + (k & 1)] = (bf16)xb[e];
  }
  if (tid < MM) {
    float s = 0.f;
    for (int k = 0; k < KK; ++k) s += ab[k * MM + tid];
    outPres[(size_t)bt * MM + tid] = (s > 0.001f) ? 1.0f : 0.0f;
    rden[tid] = 1.0f / fmaxf(s, 1e-6f);
  }
  __syncthreads();

  int lane = tid & 31, w = tid >> 5;
  int mt = w & 1, grp = w >> 1;              // wave -> m-tile (0..1), n-group (0..3)
  bf16x16 a0 = load_a_frag(At + mt * 16 * KK,      KK, lane);
  bf16x16 a1 = load_a_frag(At + mt * 16 * KK + 32, KK, lane);
  int r = lane & 15, g = lane >> 4;
#pragma unroll
  for (int j = 0; j < 6; ++j) {
    int nt = grp * 6 + j;
    f32x8 acc = {};
    bf16x16 b0 = load_b_frag_i(Xs + nt * 32,            DD, lane);  // k0=0
    WMMA_BF16(acc, a0, b0);
    bf16x16 b1 = load_b_frag_i(Xs + 32 * DD + nt * 32,  DD, lane);  // k0=32
    WMMA_BF16(acc, a1, b1);
    int n = nt * 16 + r;
#pragma unroll
    for (int q = 0; q < 8; ++q) {
      int m = mt * 16 + q + 8 * g;
      outU[((size_t)bt * MM + m) * DD + n] = acc[q] * rden[m];
    }
  }
}

// ---------------------------------------------------------------------------
// Generic bf16-WMMA GEMM:  C[r,n] = sum_k A[r,k]*Bm[k,n] (+bias[n])
// A: f32, lda = 384 (fixed). Block tile: 64 rows x 128 cols, k-chunks of 32.
// Bm is k-pair interleaved in global; per k-chunk a 32x128 tile is 16
// contiguous 512B runs -> one async b128 per thread, double-buffered.
// Full A tile staged to LDS first => safe for in-place C==A.
// ---------------------------------------------------------------------------
__device__ inline void stage_b_gemm(bf16* dst, const bf16* Bm, int N, int kc, int n0, int tid) {
  int run = tid >> 4, q = tid & 15;          // 16 runs x 16 xfers = 256 threads
  async_copy16(dst + run * 256 + q * 8,
               Bm + (size_t)(16 * kc + run) * 2 * N + n0 * 2 + q * 8);
}

__global__ __launch_bounds__(256) void k_gemm(const float* __restrict__ A,
                                              const bf16* __restrict__ Bm,
                                              const float* __restrict__ bias,
                                              void* __restrict__ Cout, int N, int cbf16) {
  __shared__ alignas(16) bf16 As[64 * DD];
  __shared__ alignas(16) bf16 Bs[2][32 * 128];   // k-pair interleaved, 2 buffers
  int tid = threadIdx.x, lane = tid & 31, w = tid >> 5;
  size_t r0 = (size_t)blockIdx.x * 64;
  int n0 = blockIdx.y * 128;

  stage_b_gemm(Bs[0], Bm, N, 0, n0, tid);
  for (int e = tid; e < 64 * DD; e += 256) As[e] = (bf16)A[r0 * DD + e];
  int mt = w >> 1, ng = w & 1;                   // wave -> m-tile, n-group
  f32x8 acc[4];
#pragma unroll
  for (int j = 0; j < 4; ++j) acc[j] = (f32x8){};

  for (int kc = 0; kc < 12; ++kc) {
    int cur = kc & 1;
    if (kc + 1 < 12) {
      stage_b_gemm(Bs[cur ^ 1], Bm, N, kc + 1, n0, tid);
      WAIT_ASYNC(1);                             // retire current batch only
    } else {
      WAIT_ASYNC(0);
    }
    __syncthreads();
    bf16x16 af = load_a_frag(As + mt * 16 * DD + kc * 32, DD, lane);
#pragma unroll
    for (int j = 0; j < 4; ++j) {
      bf16x16 bfr = load_b_frag_i(Bs[cur] + (ng * 4 + j) * 32, 128, lane);
      WMMA_BF16(acc[j], af, bfr);
    }
    __syncthreads();
  }

  int r = lane & 15, g = lane >> 4;
#pragma unroll
  for (int j = 0; j < 4; ++j) {
    int n = n0 + (ng * 4 + j) * 16 + r;
    float bv = bias ? bias[n] : 0.f;
#pragma unroll
    for (int q = 0; q < 8; ++q) {
      size_t row = r0 + mt * 16 + q + 8 * g;
      float v = acc[j][q] + bv;
      if (cbf16) ((bf16*)Cout)[row * N + n] = (bf16)v;
      else       ((float*)Cout)[row * N + n] = v;
    }
  }
}

// ---------------------------------------------------------------------------
// GRU scan: 16 persistent blocks x 16 sequences, h kept in LDS.
// Per step: gh = h @ W_hh^T via WMMA. W_hh (bf16, interleaved) is streamed
// from L2 every step with async b128 copies, double-buffered so the next
// k-chunk is in flight while the current one feeds the matrix pipe.
// Elementwise GRU update + fused LayerNorm, z_dyn written directly.
// ---------------------------------------------------------------------------
__device__ inline void stage_b_gru(bf16* dst, const bf16* whhI, int kc, int c, int tid) {
#pragma unroll
  for (int e = tid; e < 1536; e += 256) {        // 16 runs x 96 xfers (6/thread)
    int run = e / 96, q = e - run * 96;
    async_copy16(dst + run * 768 + q * 8,
                 whhI + (size_t)(16 * kc + run) * 2304 + c * 768 + q * 8);
  }
}

__global__ __launch_bounds__(256) void k_gru(const bf16* __restrict__ gx,
                                             const bf16* __restrict__ whhI,
                                             const float* __restrict__ bhh,
                                             const float* __restrict__ gdyn,
                                             const float* __restrict__ bdyn,
                                             float* __restrict__ zdyn) {
  __shared__ float hrow[16 * DD];                 // 24 KB : h state
  __shared__ alignas(16) bf16 hbf[16 * 32];       //  1 KB : A staging
  __shared__ alignas(16) bf16 Bsh[2][32 * DD];    // 48 KB : W_hh slices, 2 bufs
  __shared__ float ghs[16 * D3];                  // 72 KB : gh
  __shared__ float bhhS[D3];
  __shared__ float gdS[DD], bdS[DD];
  int tid = threadIdx.x, lane = tid & 31, w = tid >> 5;
  int r = lane & 15, g = lane >> 4;

  for (int e = tid; e < 16 * DD; e += 256) hrow[e] = 0.f;
  for (int e = tid; e < D3; e += 256) bhhS[e] = bhh[e];
  for (int e = tid; e < DD; e += 256) { gdS[e] = gdyn[e]; bdS[e] = bdyn[e]; }
  __syncthreads();

  for (int t = 0; t < TT; ++t) {
    // ---- gh = h @ W_hh^T : 3 n-chunks of 384, 12 k-chunks of 32 ----
    for (int c = 0; c < 3; ++c) {
      f32x8 acc[3];
#pragma unroll
      for (int j = 0; j < 3; ++j) acc[j] = (f32x8){};
      stage_b_gru(Bsh[0], whhI, 0, c, tid);
      for (int kc = 0; kc < 12; ++kc) {
        int cur = kc & 1;
        if (kc + 1 < 12) stage_b_gru(Bsh[cur ^ 1], whhI, kc + 1, c, tid);
        for (int e = tid; e < 512; e += 256) {    // h -> bf16 A tile
          int i = e >> 5, kk = e & 31;
          hbf[e] = (bf16)hrow[i * DD + kc * 32 + kk];
        }
        if (kc + 1 < 12) { WAIT_ASYNC(6); } else { WAIT_ASYNC(0); }
        __syncthreads();
        bf16x16 af = load_a_frag(hbf, 32, lane);
#pragma unroll
        for (int j = 0; j < 3; ++j) {
          bf16x16 bfr = load_b_frag_i(Bsh[cur] + (w * 3 + j) * 32, DD, lane);
          WMMA_BF16(acc[j], af, bfr);
        }
        __syncthreads();
      }
#pragma unroll
      for (int j = 0; j < 3; ++j) {
        int n = c * DD + (w * 3 + j) * 16 + r;
#pragma unroll
        for (int q = 0; q < 8; ++q) ghs[(q + 8 * g) * D3 + n] = acc[j][q];
      }
    }
    __syncthreads();

    // ---- elementwise GRU update (in-place h) ----
    for (int e = tid; e < 16 * DD; e += 256) {
      int i = e / DD, d = e - i * DD;
      int seq = blockIdx.x * 16 + i;
      int b = seq >> 5, m = seq & 31;
      size_t gb = ((size_t)(b * TT + t) * MM + m) * D3;
      float xr = (float)gx[gb + d];
      float xz = (float)gx[gb + DD + d];
      float xn = (float)gx[gb + 2 * DD + d];
      float hr = ghs[i * D3 + d]          + bhhS[d];
      float hz = ghs[i * D3 + DD + d]     + bhhS[DD + d];
      float hn = ghs[i * D3 + 2 * DD + d] + bhhS[2 * DD + d];
      float rg = sigmoidf_(xr + hr);
      float zg = sigmoidf_(xz + hz);
      float ng = tanhf(xn + rg * hn);
      hrow[e] = (1.f - zg) * ng + zg * hrow[e];
    }
    __syncthreads();

    // ---- LayerNorm + write z_dyn (wave w owns rows 2w, 2w+1) ----
#pragma unroll
    for (int ii = 0; ii < 2; ++ii) {
      int i = w * 2 + ii;
      float s1 = 0.f, s2 = 0.f;
      for (int d = lane; d < DD; d += 32) {
        float v = hrow[i * DD + d];
        s1 += v; s2 += v * v;
      }
#pragma unroll
      for (int off = 16; off; off >>= 1) {
        s1 += __shfl_xor(s1, off, 32);
        s2 += __shfl_xor(s2, off, 32);
      }
      float mu = s1 * (1.f / DD);
      float var = s2 * (1.f / DD) - mu * mu;
      float rs = rsqrtf(var + 1e-5f);
      int seq = blockIdx.x * 16 + i;
      int b = seq >> 5, m = seq & 31;
      size_t ob = ((size_t)(b * TT + t) * MM + m) * DD;
      for (int d = lane; d < DD; d += 32)
        zdyn[ob + d] = (hrow[i * DD + d] - mu) * rs * gdS[d] + bdS[d];
    }
    __syncthreads();
  }
}

// ---------------------------------------------------------------------------
// Semantic path: per (b,m) block (384 threads = one lane per d).
// In-place over z_sem region: raw -> LN -> gate/alpha -> presence-gated EMA.
// ---------------------------------------------------------------------------
__global__ __launch_bounds__(384) void k_sem(float* __restrict__ zsem,
                                             const float* __restrict__ pres,
                                             const float* __restrict__ gsem,
                                             const float* __restrict__ bsem,
                                             const float* __restrict__ gatew,
                                             const float* __restrict__ gateb,
                                             float* __restrict__ alpha) {
  __shared__ float red[32];
  int tid = threadIdx.x, lane = tid & 31, w = tid >> 5;
  int seq = blockIdx.x, b = seq >> 5, m = seq & 31;
  float gw = gatew[tid], gg = gsem[tid], gb = bsem[tid], g0 = gateb[0];
  float prev = 0.f;

  for (int t = 0; t < TT; ++t) {
    size_t base = ((size_t)(b * TT + t) * MM + m) * DD;
    float raw = zsem[base + tid];
    float s1 = raw, s2 = raw * raw;
#pragma unroll
    for (int off = 16; off; off >>= 1) {
      s1 += __shfl_xor(s1, off, 32);
      s2 += __shfl_xor(s2, off, 32);
    }
    if (!lane) { red[w] = s1; red[w + 16] = s2; }
    __syncthreads();
    if (!tid) {
      float a = 0.f, c = 0.f;
      for (int i = 0; i < 12; ++i) { a += red[i]; c += red[i + 16]; }
      red[0] = a; red[16] = c;
    }
    __syncthreads();
    float mu = red[0] * (1.f / DD);
    float var = red[16] * (1.f / DD) - mu * mu;
    __syncthreads();

    float sem = (raw - mu) * rsqrtf(var + 1e-5f) * gg + gb;

    float gp = sem * gw;
#pragma unroll
    for (int off = 16; off; off >>= 1) gp += __shfl_xor(gp, off, 32);
    if (!lane) red[w] = gp;
    __syncthreads();
    if (!tid) {
      float a = 0.f;
      for (int i = 0; i < 12; ++i) a += red[i];
      red[0] = a;
    }
    __syncthreads();
    float al = 0.02f + sigmoidf_(red[0] + g0) * 0.10f;
    __syncthreads();

    if (!tid) alpha[(size_t)(b * TT + t) * MM + m] = al;
    float p = pres[(size_t)(b * TT + t) * MM + m];
    if (t == 0)          prev = sem;
    else if (p > 0.5f)   prev = (1.f - al) * prev + al * sem;
    zsem[base + tid] = prev;
  }
}

// ---------------------------------------------------------------------------
// Launch
// ---------------------------------------------------------------------------
extern "C" void kernel_launch(void* const* d_in, const int* in_sizes, int n_in,
                              void* d_out, int out_size, void* d_ws, size_t ws_size,
                              hipStream_t stream) {
  (void)in_sizes; (void)n_in; (void)out_size; (void)ws_size;
  const float* tf     = (const float*)d_in[0];
  const float* assign = (const float*)d_in[1];
  const float* wp     = (const float*)d_in[2];
  const float* bp     = (const float*)d_in[3];
  const float* wih    = (const float*)d_in[4];
  const float* whh    = (const float*)d_in[5];
  const float* bih    = (const float*)d_in[6];
  const float* bhh    = (const float*)d_in[7];
  const float* wsem   = (const float*)d_in[8];
  const float* bsemp  = (const float*)d_in[9];
  const float* gatew  = (const float*)d_in[10];
  const float* gateb  = (const float*)d_in[11];
  const float* gdyn   = (const float*)d_in[12];
  const float* bdyn   = (const float*)d_in[13];
  const float* gsem   = (const float*)d_in[14];
  const float* bsemn  = (const float*)d_in[15];

  float* out      = (float*)d_out;
  float* outUnit  = out + OFF_UNIT;
  float* outPres  = out + OFF_PRES;
  float* outZdyn  = out + OFF_ZDYN;
  float* outZsem  = out + OFF_ZSEM;
  float* outAlpha = out + OFF_ALPHA;

  char* ws = (char*)d_ws;                       // ~153.4 MB total
  bf16* wpT   = (bf16*)(ws);                    //   294,912 B (interleaved)
  bf16* wsemT = (bf16*)(ws + 294912);           //   294,912 B (interleaved)
  bf16* wihT  = (bf16*)(ws + 589824);           //   884,736 B (interleaved)
  bf16* whhT  = (bf16*)(ws + 1474560);          //   884,736 B (interleaved)
  bf16* gxbuf = (bf16*)(ws + 2359296);          // 150,994,944 B (bf16 gx)

  k_wtranspose_i<<<576,  256, 0, stream>>>(wp,   wpT,   DD);
  k_wtranspose_i<<<576,  256, 0, stream>>>(wsem, wsemT, DD);
  k_wtranspose_i<<<1728, 256, 0, stream>>>(wih,  wihT,  D3);
  k_wtranspose_i<<<1728, 256, 0, stream>>>(whh,  whhT,  D3);

  k_unit_pool<<<BT, 256, 0, stream>>>(tf, assign, outUnit, outPres);

  // unit projection (in place), gx, sem_proj (raw into z_sem region)
  k_gemm<<<dim3(ROWS / 64, 3), 256, 0, stream>>>(outUnit, wpT,   bp,    (void*)outUnit, DD, 0);
  k_gemm<<<dim3(ROWS / 64, 9), 256, 0, stream>>>(outUnit, wihT,  bih,   (void*)gxbuf,   D3, 1);
  k_gemm<<<dim3(ROWS / 64, 3), 256, 0, stream>>>(outUnit, wsemT, bsemp, (void*)outZsem, DD, 0);

  k_gru<<<16, 256, 0, stream>>>(gxbuf, whhT, bhh, gdyn, bdyn, outZdyn);
  k_sem<<<BB * MM, 384, 0, stream>>>(outZsem, outPres, gsem, bsemn, gatew, gateb, outAlpha);
}
